// RelPartialLearnableMultiHeadAttn_88149908783416
// MI455X (gfx1250) — matrix operand order repaired and checked by reading, not compiled, not run on hardware
//
#include <hip/hip_runtime.h>

// ---------------------------------------------------------------------------
// Transformer-XL relative multi-head attention for MI455X (gfx1250, wave32).
// Matrix math on v_wmma_f32_16x16x32_bf16 (fp32 accumulate).
// Flash-style attention: scores never hit HBM; rel_shift via 16x48 LDS band.
// Tile staging uses global_load_async_to_lds_b128 (ASYNCcnt) where possible;
// fragments are built from LDS as contiguous b128 pairs.
// ---------------------------------------------------------------------------

typedef __attribute__((ext_vector_type(16))) __bf16 v16bf;
typedef __attribute__((ext_vector_type(8)))  float  v8f;

#define NHEAD    16
#define DHEAD    64
#define DMODEL   1024
#define BSZ      4
#define QLEN     1024
#define MEMLEN   1024
#define KLEN     2048
#define HEADS_LD 3072   // qkv concatenated row

__device__ __forceinline__ __bf16 f2bf(float f) { return (__bf16)f; }
__device__ __forceinline__ float  bf2f(__bf16 b) { return (float)b; }
__device__ __forceinline__ unsigned pk2(float a, float b) {
  unsigned short lo = __builtin_bit_cast(unsigned short, (__bf16)a);
  unsigned short hi = __builtin_bit_cast(unsigned short, (__bf16)b);
  return (unsigned)lo | ((unsigned)hi << 16);
}
__device__ __forceinline__ v8f wmma_bf16(v16bf a, v16bf b, v8f c) {
  return __builtin_amdgcn_wmma_f32_16x16x32_bf16(false, a, false, b,
                                                 (short)0, c, false, false);
}

// Async direct global->LDS copy, 16B per lane (GLOBAL_LOAD_ASYNC_TO_LDS_B128,
// tracked by ASYNCcnt). LDS address = low 32 bits of the flat pointer.
__device__ __forceinline__ void async_load_b128(void* lds_ptr,
                                                const void* gptr) {
  unsigned lds = (unsigned)(unsigned long long)lds_ptr;
  asm volatile("global_load_async_to_lds_b128 %0, %1, off"
               :
               : "v"(lds), "v"((unsigned long long)gptr)
               : "memory");
}
__device__ __forceinline__ void wait_async() {
  asm volatile("s_wait_asynccnt 0x0" ::: "memory");
}

// ---------------------------------------------------------------------------
// fp32 -> bf16 convert, 4 elems/thread-step (b128 in, b64 out)
// ---------------------------------------------------------------------------
__global__ void bf16_convert_kernel(const float* __restrict__ src,
                                    __bf16* __restrict__ dst, long n) {
  long n4 = n >> 2;
  long i = (long)blockIdx.x * blockDim.x + threadIdx.x;
  long stride = (long)gridDim.x * blockDim.x;
  for (; i < n4; i += stride) {
    float4 f = ((const float4*)src)[i];
    uint2 p;
    p.x = pk2(f.x, f.y);
    p.y = pk2(f.z, f.w);
    ((uint2*)dst)[i] = p;
  }
}

// cat = concat([mem, w], axis=1) -> bf16 [BSZ, KLEN, DMODEL]
__global__ void concat_convert_kernel(const float* __restrict__ mem,
                                      const float* __restrict__ w,
                                      __bf16* __restrict__ cat) {
  const long n4 = (long)BSZ * KLEN * (DMODEL / 4);
  long i = (long)blockIdx.x * blockDim.x + threadIdx.x;
  long stride = (long)gridDim.x * blockDim.x;
  for (; i < n4; i += stride) {
    long b   = i / ((long)KLEN * 256);
    long rem = i - b * (long)KLEN * 256;
    long j   = rem >> 8;
    long c4  = rem & 255;
    float4 f = (j < MEMLEN)
                   ? ((const float4*)mem)[(b * MEMLEN + j) * 256 + c4]
                   : ((const float4*)w)[(b * QLEN + (j - MEMLEN)) * 256 + c4];
    uint2 p;
    p.x = pk2(f.x, f.y);
    p.y = pk2(f.z, f.w);
    ((uint2*)cat)[i] = p;
  }
}

// qw = q + r_w_bias, qr = q + r_r_bias  (bf16, [BSZ, QLEN, NHEAD*DHEAD])
__global__ void split_bias_kernel(const __bf16* __restrict__ heads,
                                  const float* __restrict__ rwb,
                                  const float* __restrict__ rrb,
                                  __bf16* __restrict__ qw,
                                  __bf16* __restrict__ qr) {
  const long n4 = (long)BSZ * QLEN * 256;
  long i = (long)blockIdx.x * blockDim.x + threadIdx.x;
  long stride = (long)gridDim.x * blockDim.x;
  for (; i < n4; i += stride) {
    long b   = i >> 18;                 // / (QLEN*256)
    long rem = i & ((1L << 18) - 1);
    long row = rem >> 8;
    long c4  = rem & 255;
    uint2 q2 = ((const uint2*)heads)[((b * KLEN) + MEMLEN + row) *
                                         (HEADS_LD / 4) + c4];
    float q0 = bf2f(__builtin_bit_cast(__bf16, (unsigned short)(q2.x & 0xFFFF)));
    float q1 = bf2f(__builtin_bit_cast(__bf16, (unsigned short)(q2.x >> 16)));
    float q2v = bf2f(__builtin_bit_cast(__bf16, (unsigned short)(q2.y & 0xFFFF)));
    float q3 = bf2f(__builtin_bit_cast(__bf16, (unsigned short)(q2.y >> 16)));
    float4 wb = ((const float4*)rwb)[c4];
    float4 rb = ((const float4*)rrb)[c4];
    uint2 pw, pr;
    pw.x = pk2(q0 + wb.x, q1 + wb.y);
    pw.y = pk2(q2v + wb.z, q3 + wb.w);
    pr.x = pk2(q0 + rb.x, q1 + rb.y);
    pr.y = pk2(q2v + rb.z, q3 + rb.w);
    ((uint2*)qw)[i] = pw;
    ((uint2*)qr)[i] = pr;
  }
}

// ---------------------------------------------------------------------------
// bf16 WMMA GEMM:  C[M,N] = A[M,K] * B[N,K]^T
// Block = 128 threads (4 waves), tile 64x64, k-step 32.
// bf16 operands staged via async global->LDS b128; fp32 A converted in-regs.
// ---------------------------------------------------------------------------
template <int A_BF16, int C_F32>
__global__ __launch_bounds__(128) void gemm_bf16_kernel(
    const void* __restrict__ Ap, const __bf16* __restrict__ Bw,
    void* __restrict__ Cout, int M, int N, int K) {
  __shared__ __align__(16) __bf16 As[64][32];
  __shared__ __align__(16) __bf16 Bs[64][32];
  const int lane = threadIdx.x & 31;
  const int wave = threadIdx.x >> 5;
  const int l16  = lane & 15;
  const int hiA  = (lane >= 16) ? 8 : 0;    // A-layout K offset
  const int hiB  = (lane >= 16) ? 16 : 0;   // B-layout K offset
  const int hi8  = (lane >= 16) ? 8 : 0;    // C-layout row offset
  const long m0  = (long)blockIdx.y * 64;
  const long n0  = (long)blockIdx.x * 64;

  v8f acc[4] = {};
  for (int k0 = 0; k0 < K; k0 += 32) {
    if (k0 + 32 < K) {  // hint next tiles toward GL2 (global_prefetch_b8)
      if (A_BF16)
        __builtin_prefetch(&((const __bf16*)Ap)[m0 * K + k0 + 32], 0, 1);
      else
        __builtin_prefetch(&((const float*)Ap)[m0 * K + k0 + 32], 0, 1);
    }
    if (A_BF16) {
      const __bf16* Ab = (const __bf16*)Ap;
#pragma unroll
      for (int i = 0; i < 2; ++i) {          // 256 chunks of 8 bf16
        int chunk = threadIdx.x + 128 * i;
        int r = chunk >> 2, c = (chunk & 3) * 8;
        async_load_b128(&As[r][c], &Ab[(m0 + r) * K + k0 + c]);
      }
    } else {
      const float* Af = (const float*)Ap;
#pragma unroll
      for (int i = 0; i < 4; ++i) {          // 512 chunks of 4 f32 -> bf16
        int chunk = threadIdx.x + 128 * i;
        int r = chunk >> 3, c = (chunk & 7) * 4;
        float4 f = *(const float4*)&Af[(m0 + r) * K + k0 + c];
        uint2 p;
        p.x = pk2(f.x, f.y);
        p.y = pk2(f.z, f.w);
        *(uint2*)&As[r][c] = p;
      }
    }
#pragma unroll
    for (int i = 0; i < 2; ++i) {
      int chunk = threadIdx.x + 128 * i;
      int r = chunk >> 2, c = (chunk & 3) * 8;
      async_load_b128(&Bs[r][c], &Bw[(n0 + r) * K + k0 + c]);
    }
    wait_async();
    __syncthreads();
    v16bf af;
    {
      int r = wave * 16 + l16;
#pragma unroll
      for (int v = 0; v < 8; ++v) {
        int k = 2 * v + hiB;
        af[2 * v]     = As[r][k];
        af[2 * v + 1] = As[r][k + 1];
      }
    }
#pragma unroll
    for (int nt = 0; nt < 4; ++nt) {
      v16bf bfv;
      int cN = nt * 16 + l16;
#pragma unroll
      for (int v = 0; v < 8; ++v) {
        int k = 2 * v + hiB;
        bfv[2 * v]     = Bs[cN][k];
        bfv[2 * v + 1] = Bs[cN][k + 1];
      }
      acc[nt] = wmma_bf16(af, bfv, acc[nt]);
    }
    __syncthreads();
  }
#pragma unroll
  for (int nt = 0; nt < 4; ++nt) {
    long n = n0 + nt * 16 + l16;
#pragma unroll
    for (int v = 0; v < 8; ++v) {
      long m = m0 + wave * 16 + v + hi8;
      if (C_F32) ((float*)Cout)[m * N + n] = acc[nt][v];
      else       ((__bf16*)Cout)[m * N + n] = f2bf(acc[nt][v]);
    }
  }
}

// ---------------------------------------------------------------------------
// Fused flash attention with in-tile rel_shift.
// One wave32 per (batch, head, 16-row q tile); 32 k-cols per step.
// AC: 4 WMMA, BD: 6 WMMA (aligned 48-wide rel band), PV: 4 WMMA per step.
// K/rk tiles staged via async global->LDS b128 (register fallback for OOB
// tail tiles, uniform branch); V staged transposed through registers.
// ---------------------------------------------------------------------------
__global__ __launch_bounds__(32) void attn_kernel(
    const __bf16* __restrict__ qw,    // [B,QLEN,1024]  q + r_w_bias
    const __bf16* __restrict__ qr,    // [B,QLEN,1024]  q + r_r_bias
    const __bf16* __restrict__ heads, // [B,KLEN,3072]  K at +1024, V at +2048
    const __bf16* __restrict__ rk,    // [KLEN,1024]
    __bf16* __restrict__ vec) {       // [B,QLEN,1024]
  __shared__ __align__(16) __bf16 sK[32][64];   // K tile [j][d]; also Q stage
  __shared__ __align__(16) __bf16 sVt[64][32];  // V tile transposed [d][j]
  __shared__ __align__(16) __bf16 sR[48][64];   // rk band [jp][d]
  __shared__ float  bdl[16][48];                // Braw band (rel_shift source)
  __shared__ __align__(16) __bf16 pls[16][32];  // P: C-layout -> A-layout

  const int lane = threadIdx.x & 31;
  const int i0 = blockIdx.x * 16;
  const int h  = blockIdx.y;
  const int b  = blockIdx.z;
  const int l16 = lane & 15;
  const int hi8 = (lane >= 16) ? 8 : 0;
  const int hiA = (lane >= 16) ? 8 : 0;
  const int hiB = (lane >= 16) ? 16 : 0;
  const float scale = 0.125f;  // 1/sqrt(64)

  // ---- stage Q rows (qw -> sK[0..15], qr -> sK[16..31]) via async copy ----
#pragma unroll
  for (int i = 0; i < 4; ++i) {
    int chunk = lane + 32 * i;              // 0..127 : 16 rows x 8 chunks
    int rr = chunk >> 3, c = (chunk & 7) * 8;
    long g = ((long)b * QLEN + i0 + rr) * DMODEL + h * DHEAD + c;
    async_load_b128(&sK[rr][c], &qw[g]);
    async_load_b128(&sK[16 + rr][c], &qr[g]);
  }
  wait_async();
  __syncthreads();
  v16bf qwA[2], qrA[2];
#pragma unroll
  for (int half = 0; half < 2; ++half) {
#pragma unroll
    for (int v = 0; v < 8; ++v) {
      int k = 2 * v + hiB + half * 32;
      qwA[half][2 * v]     = sK[l16][k];
      qwA[half][2 * v + 1] = sK[l16][k + 1];
      qrA[half][2 * v]     = sK[16 + l16][k];
      qrA[half][2 * v + 1] = sK[16 + l16][k + 1];
    }
  }
  __syncthreads();

  v8f oacc[4] = {};
  float mrow[8], lrow[8];
#pragma unroll
  for (int v = 0; v < 8; ++v) { mrow[v] = -3e38f; lrow[v] = 0.f; }

  const int npair = (i0 + MEMLEN + 16 + 31) / 32;
  for (int jj = 0; jj < npair; ++jj) {
    const int j0 = jj * 32;
    const int jpb = j0 - i0 + (QLEN - 16);  // aligned rel band base, >= 0

    // ---- stage K tile (async when fully in-bounds; uniform branch) ----
    if (j0 + 32 <= KLEN) {
#pragma unroll
      for (int i = 0; i < 8; ++i) {
        int chunk = lane + 32 * i;          // 0..255 : 32 rows x 8 chunks
        int jr = chunk >> 3, c = (chunk & 7) * 8;
        async_load_b128(
            &sK[jr][c],
            &heads[((long)b * KLEN + j0 + jr) * HEADS_LD + DMODEL +
                   h * DHEAD + c]);
      }
    } else {
#pragma unroll
      for (int i = 0; i < 8; ++i) {
        int chunk = lane + 32 * i;
        int jr = chunk >> 3, c = (chunk & 7) * 8;
        int j = j0 + jr;
        bool jv = (j < KLEN);
        uint4 kv = *(const uint4*)&heads[((long)b * KLEN + (jv ? j : 0)) *
                                             HEADS_LD + DMODEL + h * DHEAD + c];
        if (!jv) kv = make_uint4(0, 0, 0, 0);
        *(uint4*)&sK[jr][c] = kv;
      }
    }
    // ---- stage V tile transposed (register path: needs transpose) ----
#pragma unroll
    for (int i = 0; i < 8; ++i) {
      int chunk = lane + 32 * i;
      int jr = chunk >> 3, c = (chunk & 7) * 8;
      int j = j0 + jr;
      bool jv = (j < KLEN);
      uint4 vv = *(const uint4*)&heads[((long)b * KLEN + (jv ? j : 0)) *
                                           HEADS_LD + 2 * DMODEL + h * DHEAD +
                                           c];
      if (!jv) vv = make_uint4(0, 0, 0, 0);
      const __bf16* vp = (const __bf16*)&vv;
#pragma unroll
      for (int e = 0; e < 8; ++e) sVt[c + e][jr] = vp[e];
    }
    // ---- stage rk band (async when fully in-bounds) ----
    if (jpb + 48 <= KLEN) {
#pragma unroll
      for (int i = 0; i < 12; ++i) {
        int chunk = lane + 32 * i;          // 0..383 : 48 rows x 8 chunks
        int pr = chunk >> 3, c = (chunk & 7) * 8;
        async_load_b128(&sR[pr][c],
                        &rk[(long)(jpb + pr) * DMODEL + h * DHEAD + c]);
      }
    } else {
#pragma unroll
      for (int i = 0; i < 12; ++i) {
        int chunk = lane + 32 * i;
        int pr = chunk >> 3, c = (chunk & 7) * 8;
        int jp = jpb + pr;
        bool pv = (jp < KLEN);
        uint4 rv =
            *(const uint4*)&rk[(long)(pv ? jp : 0) * DMODEL + h * DHEAD + c];
        if (!pv) rv = make_uint4(0, 0, 0, 0);
        *(uint4*)&sR[pr][c] = rv;
      }
    }
    wait_async();
    __syncthreads();

    // ---- AC = (q + r_w_bias) . K^T ----
    v8f s[2] = {};
#pragma unroll
    for (int ct = 0; ct < 2; ++ct) {
      int jr = ct * 16 + l16;
#pragma unroll
      for (int half = 0; half < 2; ++half) {
        v16bf kb;
#pragma unroll
        for (int v = 0; v < 8; ++v) {
          int k = 2 * v + hiB + half * 32;
          kb[2 * v]     = sK[jr][k];
          kb[2 * v + 1] = sK[jr][k + 1];
        }
        s[ct] = wmma_bf16(qwA[half], kb, s[ct]);
      }
    }

    // ---- BD band: 3 Braw tiles ----
#pragma unroll
    for (int tt = 0; tt < 3; ++tt) {
      int pr = tt * 16 + l16;
      v8f t = {};
#pragma unroll
      for (int half = 0; half < 2; ++half) {
        v16bf rb;
#pragma unroll
        for (int v = 0; v < 8; ++v) {
          int k = 2 * v + hiB + half * 32;
          rb[2 * v]     = sR[pr][k];
          rb[2 * v + 1] = sR[pr][k + 1];
        }
        t = wmma_bf16(qrA[half], rb, t);
      }
#pragma unroll
      for (int v = 0; v < 8; ++v) bdl[v + hi8][tt * 16 + l16] = t[v];
    }
    __syncthreads();

    // ---- combine + rel_shift gather + mask + online softmax ----
#pragma unroll
    for (int v = 0; v < 8; ++v) {
      const int il = v + hi8;
      const int irow = i0 + il;
      float mx = -3e38f;
#pragma unroll
      for (int ct = 0; ct < 2; ++ct) {
        int j = j0 + ct * 16 + l16;
        int jp_loc = ct * 16 + l16 + 15 - il;  // 0..46
        float sv = (s[ct][v] + bdl[il][jp_loc]) * scale;
        if (j > irow + MEMLEN) sv = -3e38f;    // mask (also kills j >= KLEN)
        s[ct][v] = sv;
        mx = fmaxf(mx, sv);
      }
#pragma unroll
      for (int off = 8; off >= 1; off >>= 1)
        mx = fmaxf(mx, __shfl_xor(mx, off, 16));
      float mnew = fmaxf(mrow[v], mx);
      float corr = __expf(mrow[v] - mnew);
      mrow[v] = mnew;
      lrow[v] *= corr;
#pragma unroll
      for (int nt = 0; nt < 4; ++nt) oacc[nt][v] *= corr;
      float ps = 0.f;
#pragma unroll
      for (int ct = 0; ct < 2; ++ct) {
        float p = __expf(s[ct][v] - mnew);
        s[ct][v] = p;
        ps += p;
      }
#pragma unroll
      for (int off = 8; off >= 1; off >>= 1) ps += __shfl_xor(ps, off, 16);
      lrow[v] += ps;
      pls[il][l16]      = f2bf(s[0][v]);
      pls[il][16 + l16] = f2bf(s[1][v]);
    }
    __syncthreads();

    // ---- P x V (P re-read in A layout; V frags from transposed LDS) ----
    v16bf pa;
#pragma unroll
    for (int v = 0; v < 8; ++v) {
      int k = 2 * v + hiB;
      pa[2 * v]     = pls[l16][k];
      pa[2 * v + 1] = pls[l16][k + 1];
    }
#pragma unroll
    for (int nt = 0; nt < 4; ++nt) {
      v16bf vb;
      int dr = nt * 16 + l16;
#pragma unroll
      for (int v = 0; v < 8; ++v) {
        int kk = 2 * v + hiB;
        vb[2 * v]     = sVt[dr][kk];
        vb[2 * v + 1] = sVt[dr][kk + 1];
      }
      oacc[nt] = wmma_bf16(pa, vb, oacc[nt]);
    }
    __syncthreads();
  }

  // ---- epilogue: vec = O / l, staged through LDS for b128 writeback ----
#pragma unroll
  for (int nt = 0; nt < 4; ++nt)
#pragma unroll
    for (int v = 0; v < 8; ++v)
      sK[v + hi8][nt * 16 + l16] = f2bf(oacc[nt][v] / lrow[v]);
  __syncthreads();
#pragma unroll
  for (int i = 0; i < 4; ++i) {
    int chunk = lane + 32 * i;              // 0..127 : 16 rows x 8 chunks
    int rr = chunk >> 3, c = (chunk & 7) * 8;
    *(uint4*)&vec[((long)b * QLEN + i0 + rr) * DMODEL + h * DHEAD + c] =
        *(const uint4*)&sK[rr][c];
  }
}

// ---------------------------------------------------------------------------
// Residual + LayerNorm: out = LN(w + attn_out); one row per block, float4 I/O
// ---------------------------------------------------------------------------
__global__ __launch_bounds__(256) void ln_kernel(
    const float* __restrict__ w, const float* __restrict__ attn,
    const float* __restrict__ gamma, const float* __restrict__ beta,
    float* __restrict__ out) {
  __shared__ float sbuf[256], s2buf[256];
  long row = blockIdx.x;
  const float4* xw = (const float4*)(w + row * DMODEL);
  const float4* xa = (const float4*)(attn + row * DMODEL);
  float4 a = xw[threadIdx.x];
  float4 c = xa[threadIdx.x];
  float x0 = a.x + c.x, x1 = a.y + c.y, x2 = a.z + c.z, x3 = a.w + c.w;
  sbuf[threadIdx.x]  = x0 + x1 + x2 + x3;
  s2buf[threadIdx.x] = x0 * x0 + x1 * x1 + x2 * x2 + x3 * x3;
  __syncthreads();
  for (int off = 128; off > 0; off >>= 1) {
    if ((int)threadIdx.x < off) {
      sbuf[threadIdx.x]  += sbuf[threadIdx.x + off];
      s2buf[threadIdx.x] += s2buf[threadIdx.x + off];
    }
    __syncthreads();
  }
  float mu = sbuf[0] * (1.f / DMODEL);
  float var = s2buf[0] * (1.f / DMODEL) - mu * mu;
  float rs = rsqrtf(var + 1e-5f);
  float4 g = ((const float4*)gamma)[threadIdx.x];
  float4 be = ((const float4*)beta)[threadIdx.x];
  float4 o;
  o.x = (x0 - mu) * rs * g.x + be.x;
  o.y = (x1 - mu) * rs * g.y + be.y;
  o.z = (x2 - mu) * rs * g.z + be.z;
  o.w = (x3 - mu) * rs * g.w + be.w;
  ((float4*)(out + row * DMODEL))[threadIdx.x] = o;
}

// ---------------------------------------------------------------------------
extern "C" void kernel_launch(void* const* d_in, const int* in_sizes, int n_in,
                              void* d_out, int out_size, void* d_ws,
                              size_t ws_size, hipStream_t stream) {
  const float* w    = (const float*)d_in[0];
  const float* r    = (const float*)d_in[1];
  const float* mem  = (const float*)d_in[2];
  // d_in[3] attention_mask: implicit (j > i + MEMLEN)
  const float* qkvw = (const float*)d_in[4];
  const float* rw   = (const float*)d_in[5];
  const float* ow   = (const float*)d_in[6];
  const float* rrb  = (const float*)d_in[7];
  const float* rwb  = (const float*)d_in[8];
  const float* gam  = (const float*)d_in[9];
  const float* bet  = (const float*)d_in[10];

  char* ws = (char*)d_ws;
  __bf16* qkvw_b = (__bf16*)(ws);                 //  6 MB
  __bf16* rw_b   = (__bf16*)(ws + 6291456);       //  2 MB
  __bf16* ow_b   = (__bf16*)(ws + 8388608);       //  2 MB
  __bf16* cat_b  = (__bf16*)(ws + 10485760);      // 16 MB
  __bf16* heads  = (__bf16*)(ws + 27262976);      // 48 MB
  __bf16* rk_b   = (__bf16*)(ws + 77594624);      //  4 MB
  __bf16* qw_b   = (__bf16*)(ws + 81788928);      //  8 MB
  __bf16* qr_b   = (__bf16*)(ws + 90177536);      //  8 MB
  __bf16* vec_b  = (__bf16*)(ws + 98566144);      //  8 MB
  float*  attn_f = (float*)(ws + 106954752);      // 16 MB (total ~118 MB)

  bf16_convert_kernel<<<dim3(512), 256, 0, stream>>>(qkvw, qkvw_b,
                                                     3L * 1024 * 1024);
  bf16_convert_kernel<<<dim3(256), 256, 0, stream>>>(rw, rw_b, 1L << 20);
  bf16_convert_kernel<<<dim3(256), 256, 0, stream>>>(ow, ow_b, 1L << 20);
  concat_convert_kernel<<<dim3(1024), 256, 0, stream>>>(mem, w, cat_b);

  // heads = cat x qkv_w^T  (8192 x 3072 x 1024)
  gemm_bf16_kernel<1, 0><<<dim3(48, 128), 128, 0, stream>>>(
      cat_b, qkvw_b, heads, 8192, 3072, 1024);
  // rk = r x r_w^T         (2048 x 1024 x 1024)
  gemm_bf16_kernel<0, 0><<<dim3(16, 32), 128, 0, stream>>>(
      r, rw_b, rk_b, 2048, 1024, 1024);
  split_bias_kernel<<<dim3(1024), 256, 0, stream>>>(heads, rwb, rrb, qw_b,
                                                    qr_b);
  // fused flash attention with rel_shift
  attn_kernel<<<dim3(QLEN / 16, NHEAD, BSZ), 32, 0, stream>>>(qw_b, qr_b,
                                                              heads, rk_b,
                                                              vec_b);
  // attn_out = vec x o_w^T (4096 x 1024 x 1024), fp32 out
  gemm_bf16_kernel<1, 1><<<dim3(16, 64), 128, 0, stream>>>(
      vec_b, ow_b, attn_f, 4096, 1024, 1024);
  ln_kernel<<<dim3(4096), 256, 0, stream>>>(w, attn_f, gam, bet,
                                            (float*)d_out);
}
